// Net_3607772529378
// MI455X (gfx1250) — matrix operand order repaired and checked by reading, not compiled, not run on hardware
//
#include <hip/hip_runtime.h>
#include <hip/hip_bf16.h>

#define BETA 0.95f
#define THR  1.0f
#define NUM_STEPS 5

#define B_DIM 4096
#define NIN   2048
#define NH    1024
#define NOUT  512

typedef __attribute__((ext_vector_type(16))) _Float16 v16h;
typedef __attribute__((ext_vector_type(8)))  _Float16 v8h;
typedef __attribute__((ext_vector_type(8)))  float    v8f;

union Frag16 {
    v16h v;
    v8h  h[2];
};

// ---------------------------------------------------------------------------
// A-matrix fragment (16x32 f16, M x K), rows m0..m0+15, cols k0..k0+31.
// Lane l<16: row=l, K=0..15 in elems 0..15.
// Lane l>=16: row=l-16, K=16..31.
// ---------------------------------------------------------------------------
__device__ inline v16h load_a_frag(const _Float16* __restrict__ base, int lda,
                                   int m0, int k0, int lane) {
    int r    = lane & 15;
    int hsel = lane >> 4;
    const _Float16* p = base + (size_t)(m0 + r) * lda + k0 + hsel * 16;
    Frag16 f;
    f.h[0] = *(const v8h*)(p);
    f.h[1] = *(const v8h*)(p + 8);
    return f.v;
}

// ---------------------------------------------------------------------------
// B-matrix fragment (32x16 f16, K x N) where B = W^T and W is row-major [N,K].
// Lane l<16: col n0+l, K=0..15 (16 contiguous halves of W row). Lane l>=16:
// col n0+l-16, K=16..31. A B-column is a contiguous W row: no transpose op.
// ---------------------------------------------------------------------------
__device__ inline v16h load_b_fragT(const _Float16* __restrict__ W, int ldw,
                                    int n0, int k0, int lane) {
    int c    = lane & 15;
    int hsel = lane >> 4;
    const _Float16* p = W + (size_t)(n0 + c) * ldw + k0 + hsel * 16;
    Frag16 f;
    f.h[0] = *(const v8h*)(p);
    f.h[1] = *(const v8h*)(p + 8);
    return f.v;
}

// ---------------------------------------------------------------------------
// fc1 + LIF1 fused: cur1 = x @ W1^T + b1 (computed once); the 5-step mem1
// recurrence is elementwise in cur1 -> emit spk1_t (f16 0/1) for all t.
// One wave per 64x64 C tile: 4x4 WMMA tiles -> 16 v_wmma per K=32 step vs
// 16 b128 loads (4x fragment reuse, ~32 FLOP/byte from L2).
// ---------------------------------------------------------------------------
__global__ __launch_bounds__(32)
void fc1_lif1_kernel(const _Float16* __restrict__ x16,
                     const _Float16* __restrict__ w116,
                     const float*    __restrict__ bias1,
                     _Float16*       __restrict__ spk1) {
    const int lane = threadIdx.x & 31;
    const int m0 = blockIdx.x * 64;
    const int n0 = blockIdx.y * 64;

    v8f acc[4][4] = {};

    for (int k0 = 0; k0 < NIN; k0 += 32) {
        if (k0 + 32 < NIN) {
            __builtin_prefetch(x16  + (size_t)(m0 + (lane & 15)) * NIN + k0 + 32, 0, 3);
            __builtin_prefetch(w116 + (size_t)(n0 + (lane & 15)) * NIN + k0 + 32, 0, 3);
        }
        v16h aF[4], bF[4];
        #pragma unroll
        for (int i = 0; i < 4; ++i)
            aF[i] = load_a_frag(x16, NIN, m0 + i * 16, k0, lane);
        #pragma unroll
        for (int j = 0; j < 4; ++j)
            bF[j] = load_b_fragT(w116, NIN, n0 + j * 16, k0, lane);
        #pragma unroll
        for (int i = 0; i < 4; ++i)
            #pragma unroll
            for (int j = 0; j < 4; ++j)
                acc[i][j] = __builtin_amdgcn_wmma_f32_16x16x32_f16(
                    false, aF[i], false, bF[j], (short)0, acc[i][j], false, false);
    }

    // C layout: VGPR g, lane l -> row = g + 8*(l>>4), col = l&15.
    const int cc   = lane & 15;
    const int rsel = (lane >> 4) * 8;

    #pragma unroll
    for (int j = 0; j < 4; ++j) {
        const int colj = n0 + j * 16 + cc;
        const float bv = bias1[colj];
        #pragma unroll
        for (int i = 0; i < 4; ++i) {
            #pragma unroll
            for (int g = 0; g < 8; ++g) {
                const int row = m0 + i * 16 + rsel + g;
                const float cur1 = acc[i][j][g] + bv;
                float mem = 0.0f;
                #pragma unroll
                for (int t = 0; t < NUM_STEPS; ++t) {
                    const float reset = (mem > THR) ? THR : 0.0f; // from PREVIOUS mem
                    mem = BETA * mem + cur1 - reset;
                    spk1[(size_t)t * B_DIM * NH + (size_t)row * NH + colj] =
                        (mem > THR) ? (_Float16)1.0f : (_Float16)0.0f;
                }
            }
        }
    }
}

// ---------------------------------------------------------------------------
// fc2 + LIF2 fused across ALL 5 timesteps: one wave per 16x32 output tile
// carries 5x2 accumulator sets; each W2 B-fragment is reused by 5 WMMA and
// each spk1 A-fragment by 2 WMMA (10 v_wmma per 14 b128 loads per K-step).
// Epilogue runs the mem2 recurrence and writes spk2 + mem2 (f32).
// ---------------------------------------------------------------------------
__global__ __launch_bounds__(32)
void fc2_lif2_kernel(const _Float16* __restrict__ spk1,
                     const _Float16* __restrict__ w216,
                     const float*    __restrict__ bias2,
                     float*          __restrict__ out) {
    const int lane = threadIdx.x & 31;
    const int m0 = blockIdx.x * 16;
    const int n0 = blockIdx.y * 32;

    v8f acc[NUM_STEPS][2] = {};

    for (int k0 = 0; k0 < NH; k0 += 32) {
        if (k0 + 32 < NH) {
            __builtin_prefetch(w216 + (size_t)(n0 + (lane & 15)) * NH + k0 + 32, 0, 3);
        }
        v16h b0 = load_b_fragT(w216, NH, n0,      k0, lane);
        v16h b1 = load_b_fragT(w216, NH, n0 + 16, k0, lane);
        #pragma unroll
        for (int t = 0; t < NUM_STEPS; ++t) {
            v16h a = load_a_frag(spk1 + (size_t)t * B_DIM * NH, NH, m0, k0, lane);
            acc[t][0] = __builtin_amdgcn_wmma_f32_16x16x32_f16(
                false, a, false, b0, (short)0, acc[t][0], false, false);
            acc[t][1] = __builtin_amdgcn_wmma_f32_16x16x32_f16(
                false, a, false, b1, (short)0, acc[t][1], false, false);
        }
    }

    const int cc   = lane & 15;
    const int rsel = (lane >> 4) * 8;
    float* __restrict__ out_spk = out;
    float* __restrict__ out_mem = out + (size_t)NUM_STEPS * B_DIM * NOUT;

    #pragma unroll
    for (int j = 0; j < 2; ++j) {
        const int col = n0 + j * 16 + cc;
        const float bv = bias2[col];
        #pragma unroll
        for (int g = 0; g < 8; ++g) {
            const int row = m0 + rsel + g;
            float mem = 0.0f;
            #pragma unroll
            for (int t = 0; t < NUM_STEPS; ++t) {
                const float cur2  = acc[t][j][g] + bv;
                const float reset = (mem > THR) ? THR : 0.0f; // from PREVIOUS mem
                mem = BETA * mem + cur2 - reset;
                const size_t idx = (size_t)t * B_DIM * NOUT + (size_t)row * NOUT + col;
                out_spk[idx] = (mem > THR) ? 1.0f : 0.0f;
                out_mem[idx] = mem;
            }
        }
    }
}

// ---------------------------------------------------------------------------
// f32 -> f16 staging (x, W1, W2): halves GEMM read traffic, enables b128
// fragment loads in this file's contiguous-k (k = 16h + s) order.
// ---------------------------------------------------------------------------
__global__ void cvt_f32_to_f16(const float* __restrict__ in,
                               _Float16* __restrict__ out, int n) {
    int i = blockIdx.x * blockDim.x + threadIdx.x;
    if (i < n) out[i] = (_Float16)in[i];
}

extern "C" void kernel_launch(void* const* d_in, const int* in_sizes, int n_in,
                              void* d_out, int out_size, void* d_ws, size_t ws_size,
                              hipStream_t stream) {
    const float* x  = (const float*)d_in[0];
    const float* W1 = (const float*)d_in[1];
    const float* b1 = (const float*)d_in[2];
    const float* W2 = (const float*)d_in[3];
    const float* b2 = (const float*)d_in[4];

    // Workspace layout (f16): x | W1 | W2 | spk1[t=0..4]  (~64 MB total)
    _Float16* x16  = (_Float16*)d_ws;
    _Float16* w116 = x16  + (size_t)B_DIM * NIN;   // 8.4M halves
    _Float16* w216 = w116 + (size_t)NH    * NIN;   // 2.1M halves
    _Float16* spk1 = w216 + (size_t)NOUT  * NH;    // 5 * 4096*1024 halves

    const int nx = B_DIM * NIN;
    const int n1 = NH * NIN;
    const int n2 = NOUT * NH;
    cvt_f32_to_f16<<<(nx + 255) / 256, 256, 0, stream>>>(x,  x16,  nx);
    cvt_f32_to_f16<<<(n1 + 255) / 256, 256, 0, stream>>>(W1, w116, n1);
    cvt_f32_to_f16<<<(n2 + 255) / 256, 256, 0, stream>>>(W2, w216, n2);

    dim3 g1(B_DIM / 64, NH / 64);    // (64, 16) waves, 64x64 tiles
    fc1_lif1_kernel<<<g1, 32, 0, stream>>>(x16, w116, b1, spk1);

    dim3 g2(B_DIM / 16, NOUT / 32);  // (256, 16) waves, 16x32 tiles x 5 steps
    fc2_lif2_kernel<<<g2, 32, 0, stream>>>(spk1, w216, b2, (float*)d_out);
}
